// SFIB_42545946034709
// MI455X (gfx1250) — matrix operand order repaired and checked by reading, not compiled, not run on hardware
//
#include <hip/hip_runtime.h>
#include <hip/hip_bf16.h>

typedef __bf16 bf16;
typedef __attribute__((ext_vector_type(16))) __bf16 v16bf;
typedef __attribute__((ext_vector_type(8)))  __bf16 v8bf;
typedef __attribute__((ext_vector_type(4)))  __bf16 v4bf;
typedef __attribute__((ext_vector_type(8)))  float  v8f;

#define LDST 40   // LDS row stride in bf16 elements (80B = 5*16B: keeps 16B-aligned vector ops)

__device__ __forceinline__ v16bf cat8(v8bf a, v8bf b) {
  return __builtin_shufflevector(a, b, 0,1,2,3,4,5,6,7,8,9,10,11,12,13,14,15);
}
// A fragment 16x32 bf16 from LDS [m][k]: lane l (lo=l&15 is M, hi=l>>4): j<8 -> K=8*hi+j ; j>=8 -> K=16+8*hi+(j-8)
__device__ __forceinline__ v16bf load_afrag(const bf16* As, int row, int hi) {
  const bf16* p = As + row * LDST;
  v8bf c0 = *(const v8bf*)(p + 8 * hi);
  v8bf c1 = *(const v8bf*)(p + 16 + 8 * hi);
  return cat8(c0, c1);
}
// B fragment 32x16 bf16 from LDS stored transposed [n][k]: lane l (n=l&15, hi=l>>4): j<8 -> K=8*hi+j ; j>=8 -> K=16+8*hi+(j-8)
__device__ __forceinline__ v16bf load_bfrag(const bf16* Bs, int nrow, int hi) {
  const bf16* p = Bs + nrow * LDST;
  v8bf c0 = *(const v8bf*)(p + 8 * hi);
  v8bf c1 = *(const v8bf*)(p + 16 + 8 * hi);
  return cat8(c0, c1);
}
__device__ __forceinline__ v8f wmma_bf16(v16bf a, v16bf b, v8f c) {
  return __builtin_amdgcn_wmma_f32_16x16x32_bf16(false, a, false, b, (short)0, c, false, false);
}

// ---------------- elementwise / transform kernels ----------------
__global__ void k_fill(float* p, float v, size_t n) {
  size_t i = (size_t)blockIdx.x * blockDim.x + threadIdx.x;
  if (i < n) p[i] = v;
}
__global__ void k_cvt(const float* __restrict__ s, bf16* __restrict__ d, size_t n) {
  size_t i = (size_t)blockIdx.x * blockDim.x + threadIdx.x;
  if (i < n) d[i] = (bf16)s[i];
}
// tiled convert+transpose: dst[c][r] = (bf16)src[r][c] ; grid (rows/32, cols/32), block (32,8)
__global__ void k_cvtT(const float* __restrict__ src, bf16* __restrict__ dst, int rows, int cols) {
  __shared__ float tile[32][33];
  int r0 = blockIdx.x * 32, c0 = blockIdx.y * 32;
  int tx = threadIdx.x, ty = threadIdx.y;
  #pragma unroll
  for (int j = 0; j < 32; j += 8)
    tile[ty + j][tx] = src[(size_t)(r0 + ty + j) * cols + c0 + tx];
  __syncthreads();
  #pragma unroll
  for (int j = 0; j < 32; j += 8)
    dst[(size_t)(c0 + ty + j) * rows + r0 + tx] = (bf16)tile[tx][ty + j];
}
// zT[c][r] = (bf16)(Hm[r][c] + mask[r][c]*Hg[r][c])
__global__ void k_zT(const float* __restrict__ Hm, const float* __restrict__ Hg,
                     const float* __restrict__ mask, bf16* __restrict__ zT, int rows, int cols) {
  __shared__ float tile[32][33];
  int r0 = blockIdx.x * 32, c0 = blockIdx.y * 32;
  int tx = threadIdx.x, ty = threadIdx.y;
  #pragma unroll
  for (int j = 0; j < 32; j += 8) {
    size_t i = (size_t)(r0 + ty + j) * cols + c0 + tx;
    tile[ty + j][tx] = Hm[i] + mask[i] * Hg[i];
  }
  __syncthreads();
  #pragma unroll
  for (int j = 0; j < 32; j += 8)
    zT[(size_t)(c0 + ty + j) * rows + r0 + tx] = (bf16)tile[tx][ty + j];
}
__global__ void k_deg(const int* __restrict__ dst, float* deg, int E) {
  int i = blockIdx.x * blockDim.x + threadIdx.x;
  if (i < E) atomicAdd(&deg[dst[i]], 1.0f);
}
__global__ void k_rsqrt(float* p, int n) {
  int i = blockIdx.x * blockDim.x + threadIdx.x;
  if (i < n) p[i] = rsqrtf(p[i]);
}
__global__ void k_spatial(const float* __restrict__ aggs, const float* __restrict__ aggt,
                          const float* __restrict__ hs, const float* __restrict__ ht,
                          const float* __restrict__ dinv, const float* __restrict__ bs,
                          const float* __restrict__ bt, const float* __restrict__ xm,
                          float* __restrict__ hspat, size_t n) {
  size_t i = (size_t)blockIdx.x * blockDim.x + threadIdx.x;
  if (i >= n) return;
  int row = (int)(i >> 7), c = (int)(i & 127);
  float di2 = dinv[row] * dinv[row];
  float sv = tanhf(aggs[i] + hs[i] * di2 + bs[c]);
  float tv = aggt[i] + ht[i] * di2 + bt[c];
  hspat[i] = xm[i] * expf(sv) + tv;
}
// wave-per-edge scatter: agg[dst] += h[src] * dinv[src]*dinv[dst]
__global__ __launch_bounds__(256) void k_scatter(
    const float* __restrict__ hs, const float* __restrict__ ht,
    const float* __restrict__ dinv, const int* __restrict__ src,
    const int* __restrict__ dst, float* __restrict__ aggs, float* __restrict__ aggt, int E) {
  int wid = (blockIdx.x * 256 + threadIdx.x) >> 5;
  int lane = threadIdx.x & 31;
  if (wid >= E) return;
  int s = src[wid], d = dst[wid];
  float nrm = dinv[s] * dinv[d];
  size_t so = (size_t)s * 128 + lane * 4;
  size_t doo = (size_t)d * 128 + lane * 4;
  float4 v = *(const float4*)(hs + so);
  atomicAdd(&aggs[doo + 0], v.x * nrm);
  atomicAdd(&aggs[doo + 1], v.y * nrm);
  atomicAdd(&aggs[doo + 2], v.z * nrm);
  atomicAdd(&aggs[doo + 3], v.w * nrm);
  float4 w = *(const float4*)(ht + so);
  atomicAdd(&aggt[doo + 0], w.x * nrm);
  atomicAdd(&aggt[doo + 1], w.y * nrm);
  atomicAdd(&aggt[doo + 2], w.z * nrm);
  atomicAdd(&aggt[doo + 3], w.w * nrm);
}
// wave-per-row fusion: w = sigmoid(dot(h_sp - h_fr, W_na) + b); h_enh = h_sp + h_fr*w
__global__ __launch_bounds__(256) void k_fusion(
    const float* __restrict__ hsp, const float* __restrict__ hfr,
    const float* __restrict__ Wna, const float* __restrict__ bna,
    bf16* __restrict__ heB, bf16* __restrict__ hfB) {
  int wave = threadIdx.x >> 5, lane = threadIdx.x & 31;
  int row = blockIdx.x * 8 + wave;
  size_t o = (size_t)row * 128 + lane * 4;
  float4 sp = *(const float4*)(hsp + o);
  float4 fr = *(const float4*)(hfr + o);
  float4 wn = *(const float4*)(Wna + lane * 4);
  float p = (sp.x - fr.x) * wn.x + (sp.y - fr.y) * wn.y +
            (sp.z - fr.z) * wn.z + (sp.w - fr.w) * wn.w;
  #pragma unroll
  for (int off = 16; off > 0; off >>= 1) p += __shfl_xor(p, off, 32);
  float g = 1.0f / (1.0f + expf(-(p + bna[0])));
  v4bf e, f;
  e[0] = (bf16)(sp.x + fr.x * g); e[1] = (bf16)(sp.y + fr.y * g);
  e[2] = (bf16)(sp.z + fr.z * g); e[3] = (bf16)(sp.w + fr.w * g);
  f[0] = (bf16)fr.x; f[1] = (bf16)fr.y; f[2] = (bf16)fr.z; f[3] = (bf16)fr.w;
  *(v4bf*)(heB + o) = e;
  *(v4bf*)(hfB + o) = f;
}

// ---------------- big GEMM: O = (TRANSA ? U^T : U) @ X ; split-K via f32 atomics ----------------
// U: [n,n] f32 (bf16 on the fly). X*T: [128][n] bf16 (pre-transposed). O: [n,128] f32 (pre-zeroed).
// Double-buffered LDS pipeline: gload(k+1) -> compute(k) -> lstore(k+1) -> barrier.
template<bool TRANSA, bool DUAL>
__global__ __launch_bounds__(256) void k_gemm_u(
    const float* __restrict__ U, const bf16* __restrict__ X0T, const bf16* __restrict__ X1T,
    float* __restrict__ O0, float* __restrict__ O1, int n, int klen) {
  __shared__ bf16 As[2][128 * LDST];
  __shared__ bf16 Bs0[2][128 * LDST];
  __shared__ bf16 Bs1[DUAL ? 2 : 1][DUAL ? 128 * LDST : 8];

  const int tid = threadIdx.x;
  const int lane = tid & 31, wave = tid >> 5;
  const int wm = wave >> 1, wn = wave & 1;      // 4x2 wave grid; each wave: 32 rows x 64 cols
  const int lo = lane & 15, hi = lane >> 4;
  const int mblk = blockIdx.x * 128;
  const int kbase = blockIdx.y * klen;

  v8f zero = {};
  v8f acc0[2][4], acc1[2][4];
  #pragma unroll
  for (int i = 0; i < 2; ++i)
    #pragma unroll
    for (int j = 0; j < 4; ++j) { acc0[i][j] = zero; acc1[i][j] = zero; }

  float ra[4][4];
  v8bf rb0[2], rb1[2];

  auto gload = [&](int k0) {
    if (TRANSA) {
      // gather along k: wave keeps kf fixed, lanes sweep 32 consecutive m -> coalesced b32 loads
      const int kf = tid >> 5, ml = tid & 31;
      #pragma unroll
      for (int it = 0; it < 4; ++it) {
        const int m = it * 32 + ml;
        #pragma unroll
        for (int j = 0; j < 4; ++j)
          ra[it][j] = U[(size_t)(k0 + kf * 4 + j) * n + mblk + m];
      }
    } else {
      #pragma unroll
      for (int it = 0; it < 4; ++it) {
        int s = tid + it * 256;
        int m = s >> 3, kf = s & 7;
        float4 v = *(const float4*)(U + (size_t)(mblk + m) * n + k0 + kf * 4);
        ra[it][0] = v.x; ra[it][1] = v.y; ra[it][2] = v.z; ra[it][3] = v.w;
      }
    }
    #pragma unroll
    for (int it = 0; it < 2; ++it) {           // straight b128 copies from pre-transposed X
      int s = tid + it * 256;
      int nn = s >> 2, kc = s & 3;
      rb0[it] = *(const v8bf*)(X0T + (size_t)nn * n + k0 + kc * 8);
      if (DUAL) rb1[it] = *(const v8bf*)(X1T + (size_t)nn * n + k0 + kc * 8);
    }
  };
  auto lstore = [&](int buf) {
    if (TRANSA) {
      const int kf = tid >> 5, ml = tid & 31;
      #pragma unroll
      for (int it = 0; it < 4; ++it) {
        const int m = it * 32 + ml;
        v4bf p; p[0] = (bf16)ra[it][0]; p[1] = (bf16)ra[it][1];
                p[2] = (bf16)ra[it][2]; p[3] = (bf16)ra[it][3];
        *(v4bf*)(&As[buf][m * LDST + kf * 4]) = p;
      }
    } else {
      #pragma unroll
      for (int it = 0; it < 4; ++it) {
        int s = tid + it * 256;
        int m = s >> 3, kf = s & 7;
        v4bf p; p[0] = (bf16)ra[it][0]; p[1] = (bf16)ra[it][1];
                p[2] = (bf16)ra[it][2]; p[3] = (bf16)ra[it][3];
        *(v4bf*)(&As[buf][m * LDST + kf * 4]) = p;
      }
    }
    #pragma unroll
    for (int it = 0; it < 2; ++it) {
      int s = tid + it * 256;
      int nn = s >> 2, kc = s & 3;
      *(v8bf*)(&Bs0[buf][nn * LDST + kc * 8]) = rb0[it];
      if (DUAL) *(v8bf*)(&Bs1[buf][nn * LDST + kc * 8]) = rb1[it];
    }
  };
  auto compute = [&](int buf) {
    v16bf a0 = load_afrag(&As[buf][0], wm * 32 + lo, hi);
    v16bf a1 = load_afrag(&As[buf][0], wm * 32 + 16 + lo, hi);
    #pragma unroll
    for (int tn = 0; tn < 4; ++tn) {
      v16bf b0 = load_bfrag(&Bs0[buf][0], wn * 64 + tn * 16 + lo, hi);
      acc0[0][tn] = wmma_bf16(a0, b0, acc0[0][tn]);
      acc0[1][tn] = wmma_bf16(a1, b0, acc0[1][tn]);
      if (DUAL) {
        v16bf b1 = load_bfrag(&Bs1[buf][0], wn * 64 + tn * 16 + lo, hi);
        acc1[0][tn] = wmma_bf16(a0, b1, acc1[0][tn]);
        acc1[1][tn] = wmma_bf16(a1, b1, acc1[1][tn]);
      }
    }
  };

  const int steps = klen / 32;
  gload(kbase);
  lstore(0);
  for (int kt = 0; kt < steps; ++kt) {
    __syncthreads();
    if (kt + 1 < steps) {
      gload(kbase + (kt + 1) * 32);
      if (kt + 2 < steps)
        __builtin_prefetch(U + (TRANSA ? (size_t)(kbase + (kt + 2) * 32) * n + mblk
                                       : (size_t)mblk * n + kbase + (kt + 2) * 32), 0, 1);
    }
    compute(kt & 1);
    if (kt + 1 < steps) lstore((kt + 1) & 1);
  }

  // C/D layout: element r -> M = r + 8*hi, N = lo
  #pragma unroll
  for (int tm = 0; tm < 2; ++tm)
    #pragma unroll
    for (int tn = 0; tn < 4; ++tn)
      #pragma unroll
      for (int r = 0; r < 8; ++r) {
        int row = mblk + wm * 32 + tm * 16 + r + 8 * hi;
        int col = wn * 64 + tn * 16 + lo;
        atomicAdd(&O0[(size_t)row * 128 + col], acc0[tm][tn][r]);
        if (DUAL) atomicAdd(&O1[(size_t)row * 128 + col], acc1[tm][tn][r]);
      }
}

// ---------------- generic small GEMM: OUT = act(A0@W0 [+ A1@W1] + bias [+ extra]) ----------------
// A*: [8192,128] bf16 row-major ; W*T: [128][128] bf16 (pre-transposed, [n][k]) ; outF f32 / outB bf16.
__global__ __launch_bounds__(256) void k_small_gemm(
    const bf16* __restrict__ A0, const bf16* __restrict__ A1,
    const bf16* __restrict__ W0T, const bf16* __restrict__ W1T,
    const float* __restrict__ bias, const float* __restrict__ extra,
    float* __restrict__ outF, bf16* __restrict__ outB, int act) {
  __shared__ bf16 As[128 * LDST];
  __shared__ bf16 Ws[128 * LDST];

  const int tid = threadIdx.x;
  const int lane = tid & 31, wave = tid >> 5;
  const int wm = wave >> 1, wn = wave & 1;
  const int lo = lane & 15, hi = lane >> 4;
  const int mblk = blockIdx.x * 128;

  v8f zero = {};
  v8f acc[2][4];
  #pragma unroll
  for (int i = 0; i < 2; ++i)
    #pragma unroll
    for (int j = 0; j < 4; ++j) acc[i][j] = zero;

  const int npass = (A1 != nullptr) ? 2 : 1;
  for (int pass = 0; pass < npass; ++pass) {
    const bf16* A = pass ? A1 : A0;
    const bf16* WT = pass ? W1T : W0T;
    for (int ks = 0; ks < 4; ++ks) {
      const int k0 = ks * 32;
      #pragma unroll
      for (int it = 0; it < 4; ++it) {       // A rows: straight b64 copies
        int s = tid + it * 256;
        int m = s >> 3, kf = s & 7;
        v4bf v = *(const v4bf*)(A + (size_t)(mblk + m) * 128 + k0 + kf * 4);
        *(v4bf*)(As + m * LDST + kf * 4) = v;
      }
      #pragma unroll
      for (int it = 0; it < 2; ++it) {       // W: straight b128 copies from pre-transposed WT
        int s = tid + it * 256;
        int nn = s >> 2, kc = s & 3;
        v8bf v = *(const v8bf*)(WT + (size_t)nn * 128 + k0 + kc * 8);
        *(v8bf*)(Ws + nn * LDST + kc * 8) = v;
      }
      __syncthreads();
      v16bf a0 = load_afrag(As, wm * 32 + lo, hi);
      v16bf a1 = load_afrag(As, wm * 32 + 16 + lo, hi);
      #pragma unroll
      for (int tn = 0; tn < 4; ++tn) {
        v16bf b = load_bfrag(Ws, wn * 64 + tn * 16 + lo, hi);
        acc[0][tn] = wmma_bf16(a0, b, acc[0][tn]);
        acc[1][tn] = wmma_bf16(a1, b, acc[1][tn]);
      }
      __syncthreads();
    }
  }

  #pragma unroll
  for (int tm = 0; tm < 2; ++tm)
    #pragma unroll
    for (int tn = 0; tn < 4; ++tn)
      #pragma unroll
      for (int r = 0; r < 8; ++r) {
        int row = mblk + wm * 32 + tm * 16 + r + 8 * hi;
        int col = wn * 64 + tn * 16 + lo;
        size_t idx = (size_t)row * 128 + col;
        float v = acc[tm][tn][r];
        if (bias)  v += bias[col];
        if (extra) v += extra[idx];
        if (act == 1)      v = fmaxf(v, 0.0f);
        else if (act == 2) v = 1.0f / (1.0f + expf(-v));
        if (outF) outF[idx] = v;
        if (outB) outB[idx] = (bf16)v;
      }
}

extern "C" void kernel_launch(void* const* d_in, const int* in_sizes, int n_in,
                              void* d_out, int out_size, void* d_ws, size_t ws_size,
                              hipStream_t stream) {
  (void)n_in; (void)out_size; (void)ws_size;
  const int C = 128;
  const int N = in_sizes[0] / C;     // 8192
  const int E = in_sizes[2] / 2;     // 262144
  const size_t NC = (size_t)N * C;

  const float* x_main  = (const float*)d_in[0];
  const float* x_guide = (const float*)d_in[1];
  const int*   src     = (const int*)d_in[2];
  const int*   dst     = src + E;
  const float* U       = (const float*)d_in[3];
  const float* W_fg1   = (const float*)d_in[4];
  const float* b_fg1   = (const float*)d_in[5];
  const float* W_fg2   = (const float*)d_in[6];
  const float* b_fg2   = (const float*)d_in[7];
  const float* W_s     = (const float*)d_in[8];
  const float* b_s     = (const float*)d_in[9];
  const float* W_t     = (const float*)d_in[10];
  const float* b_t     = (const float*)d_in[11];
  const float* W_na    = (const float*)d_in[12];
  const float* b_na    = (const float*)d_in[13];
  const float* W_out   = (const float*)d_in[14];
  const float* b_out   = (const float*)d_in[15];

  char* w = (char*)d_ws;
  auto allocF = [&](size_t ne) { float* p = (float*)w; w += ne * sizeof(float); return p; };
  auto allocB = [&](size_t ne) { bf16*  p = (bf16*)w;  w += ne * sizeof(bf16);  return p; };

  float* Hm    = allocF(NC);   // zeroed (atomic targets, contiguous group of 5)
  float* Hg    = allocF(NC);
  float* hfreq = allocF(NC);
  float* aggs  = allocF(NC);
  float* aggt  = allocF(NC);
  float* deg   = allocF(N);    // init 1.0 (self-loop); becomes dinv after rsqrt
  float* maskF = allocF(NC);
  float* hsF   = allocF(NC);
  float* htF   = allocF(NC);
  float* hspat = allocF(NC);
  bf16* xmT   = allocB(NC);    // [128][N] transposed bf16 operands for the big GEMMs
  bf16* xgT   = allocB(NC);
  bf16* zT    = allocB(NC);
  bf16* HmB   = allocB(NC);    // row-major bf16 A-operands for small GEMMs
  bf16* HgB   = allocB(NC);
  bf16* h1B   = allocB(NC);
  bf16* heB   = allocB(NC);
  bf16* hfB   = allocB(NC);
  bf16* Wfg1T = allocB(2 * C * C);   // transposed weight blocks [n][k]
  bf16* Wfg2T = allocB(C * C);
  bf16* WsT   = allocB(C * C);
  bf16* WtT   = allocB(C * C);
  bf16* WoutT = allocB(2 * C * C);

  dim3 blk(256);
  dim3 tblk(32, 8);
  const int KSPLIT = 4;
  const int klen = N / KSPLIT;

  // zero atomic accumulators; deg starts at 1 (self loop)
  k_fill<<<dim3((unsigned)((5 * NC + 255) / 256)), blk, 0, stream>>>(Hm, 0.0f, 5 * NC);
  k_fill<<<dim3((unsigned)((N + 255) / 256)), blk, 0, stream>>>(deg, 1.0f, (size_t)N);

  // transposed bf16 operands (inputs + weights)
  k_cvtT<<<dim3(N / 32, C / 32), tblk, 0, stream>>>(x_main,  xmT, N, C);
  k_cvtT<<<dim3(N / 32, C / 32), tblk, 0, stream>>>(x_guide, xgT, N, C);
  k_cvtT<<<dim3(C / 32, C / 32), tblk, 0, stream>>>(W_fg1,           Wfg1T,         C, C);
  k_cvtT<<<dim3(C / 32, C / 32), tblk, 0, stream>>>(W_fg1 + C * C,   Wfg1T + C * C, C, C);
  k_cvtT<<<dim3(C / 32, C / 32), tblk, 0, stream>>>(W_fg2, Wfg2T, C, C);
  k_cvtT<<<dim3(C / 32, C / 32), tblk, 0, stream>>>(W_s,   WsT,   C, C);
  k_cvtT<<<dim3(C / 32, C / 32), tblk, 0, stream>>>(W_t,   WtT,   C, C);
  k_cvtT<<<dim3(C / 32, C / 32), tblk, 0, stream>>>(W_out,         WoutT,         C, C);
  k_cvtT<<<dim3(C / 32, C / 32), tblk, 0, stream>>>(W_out + C * C, WoutT + C * C, C, C);

  // degree -> dinv
  k_deg<<<dim3((unsigned)((E + 255) / 256)), blk, 0, stream>>>(dst, deg, E);
  k_rsqrt<<<dim3((unsigned)((N + 255) / 256)), blk, 0, stream>>>(deg, N);

  // GFT: Hm = U^T @ x_main, Hg = U^T @ x_guide (one fused pass over U, split-K atomics)
  k_gemm_u<true, true><<<dim3(N / 128, KSPLIT), blk, 0, stream>>>(U, xmT, xgT, Hm, Hg, N, klen);

  // GCN linear parts: hs = x_guide @ W_s ; ht = x_guide @ W_t  (A = xgB row-major = reuse HmB slot? no: use xg via bf16 row-major)
  // x_guide row-major bf16 for small GEMM A: reuse HgB buffer before the gate needs it
  k_cvt<<<dim3((unsigned)((NC + 255) / 256)), blk, 0, stream>>>(x_guide, HgB, NC);
  k_small_gemm<<<dim3(N / 128), blk, 0, stream>>>(HgB, nullptr, WsT, nullptr, nullptr, nullptr, hsF, nullptr, 0);
  k_small_gemm<<<dim3(N / 128), blk, 0, stream>>>(HgB, nullptr, WtT, nullptr, nullptr, nullptr, htF, nullptr, 0);

  // edge scatter-add (normalized)
  k_scatter<<<dim3((unsigned)(E / 8)), blk, 0, stream>>>(hsF, htF, deg, src, dst, aggs, aggt, E);

  // frequency gate MLP: h1 = relu([Hm,Hg]@Wfg1 + b1); mask = sigmoid(h1@Wfg2 + b2)
  k_cvt<<<dim3((unsigned)((NC + 255) / 256)), blk, 0, stream>>>(Hm, HmB, NC);
  k_cvt<<<dim3((unsigned)((NC + 255) / 256)), blk, 0, stream>>>(Hg, HgB, NC);
  k_small_gemm<<<dim3(N / 128), blk, 0, stream>>>(HmB, HgB, Wfg1T, Wfg1T + C * C, b_fg1, nullptr, nullptr, h1B, 1);
  k_small_gemm<<<dim3(N / 128), blk, 0, stream>>>(h1B, nullptr, Wfg2T, nullptr, b_fg2, nullptr, maskF, nullptr, 2);
  k_zT<<<dim3(N / 32, C / 32), tblk, 0, stream>>>(Hm, Hg, maskF, zT, N, C);

  // iGFT: h_freq = U @ z
  k_gemm_u<false, false><<<dim3(N / 128, KSPLIT), blk, 0, stream>>>(U, zT, nullptr, hfreq, nullptr, N, klen);

  // spatial path epilogue: s=tanh(agg_s + self + b_s); h_spatial = x_main*exp(s) + t
  k_spatial<<<dim3((unsigned)((NC + 255) / 256)), blk, 0, stream>>>(aggs, aggt, hsF, htF, deg, b_s, b_t, x_main, hspat, NC);

  // fusion gate + bf16 operands for final GEMM
  k_fusion<<<dim3(N / 8), blk, 0, stream>>>(hspat, hfreq, W_na, b_na, heB, hfB);

  // out = [h_enh, h_freq] @ W_out + b_out + x_main
  k_small_gemm<<<dim3(N / 128), blk, 0, stream>>>(heB, hfB, WoutT, WoutT + C * C, b_out, x_main, (float*)d_out, nullptr, 0);
}
